// AntiviralGNN_83923660964109
// MI455X (gfx1250) — matrix operand order repaired and checked by reading, not compiled, not run on hardware
//
#include <hip/hip_runtime.h>
#include <hip/hip_fp16.h>
#include <hip/hip_bf16.h>

typedef __attribute__((ext_vector_type(16))) _Float16 v16h;
typedef __attribute__((ext_vector_type(8)))  float    v8f;

#define NNODE  20000
#define NEDGE  320000
#define E2TOT  (NNODE + NEDGE)     // edges + self loops
#define NGRAPH 256
#define HID    256
#define FNODE  64
#define FEDGE  16
#define EPSV   1e-5f
#define NEGINF_KEY 0x007FFFFFu     // f2key(-inf)

static __device__ __forceinline__ unsigned f2key(float f) {
  int i = __float_as_int(f);
  return (i >= 0) ? ((unsigned)i | 0x80000000u) : ~(unsigned)i;
}
static __device__ __forceinline__ float key2f(unsigned k) {
  int i = (k & 0x80000000u) ? (int)(k & 0x7FFFFFFFu) : (int)~k;
  return __int_as_float(i);
}

// ---------------- utility kernels ----------------
__global__ void k_fill_f32(float* p, float v, int n) {
  int i = blockIdx.x * blockDim.x + threadIdx.x;
  if (i < n) p[i] = v;
}
__global__ void k_fill_u32(unsigned* p, unsigned v, int n) {
  int i = blockIdx.x * blockDim.x + threadIdx.x;
  if (i < n) p[i] = v;
}

// Pack f32 weight matrix B[K,Nc] into WMMA B-fragment order (f16):
// Bp[(((blk*tilesPerRow)+tc)*32 + lane)*16 + j], where for lane (n=lane&15,
// h=lane>>4) and j: element = B[blk*32 + 16*(j>>3) + 8h + (j&7)][tc*16+n].
// Each lane's 16-half fragment is then a single contiguous 32-byte load.
__global__ void k_pack_b(const float* __restrict__ B, _Float16* __restrict__ Bp,
                         int K, int Nc) {
  int idx = blockIdx.x * blockDim.x + threadIdx.x;
  if (idx >= K * Nc) return;
  int j    = idx & 15;
  int lane = (idx >> 4) & 31;
  int t    = idx >> 9;                 // blk * tilesPerRow + tc
  int tilesPerRow = Nc >> 4;
  int blk = t / tilesPerRow, tc = t % tilesPerRow;
  int half = lane >> 4, n = lane & 15;
  int k = blk * 32 + 16 * (j >> 3) + 8 * half + (j & 7);
  Bp[idx] = (_Float16)B[(size_t)k * Nc + tc * 16 + n];
}

// ---------------- self-loop attr (mean of incoming edge_attr) ----------------
__global__ void k_loop_acc(const int* dst, const float* eattr, float* deg, float* loopA, int E) {
  int e = blockIdx.x * blockDim.x + threadIdx.x;
  if (e >= E) return;
  int d = dst[e];
  atomicAdd(&deg[d], 1.0f);
  #pragma unroll
  for (int k = 0; k < FEDGE; ++k)
    atomicAdd(&loopA[(size_t)d * FEDGE + k], eattr[(size_t)e * FEDGE + k]);
}
__global__ void k_loop_div(float* loopA, const float* deg, int n) {
  int i = blockIdx.x * blockDim.x + threadIdx.x;
  if (i >= n) return;
  loopA[i] /= fmaxf(deg[i / FEDGE], 1.0f);
}

// ---------------- WMMA GEMM: C[M,Nc] = A[M,K](f32->f16) * B(packed f16) ----------
// One wave computes a 16x64 C strip: 4 accumulators sharing one A fragment
// (4x A reuse, 4 back-to-back independent v_wmma per K-step).
// A-fragment per lane = two contiguous 32B chunks: K in [kk+8h, +8) and [kk+16+8h, +8).
// B-fragment per lane = one contiguous 32B chunk from the packed layout.
// EPI: 0 = +bias, 1 = +bias,relu, 2 = +bias,BN,relu
template <int EPI>
__global__ void k_gemm_wmma(const float* __restrict__ A, const _Float16* __restrict__ Bp,
                            const float* __restrict__ bias,
                            const float* __restrict__ bn_g, const float* __restrict__ bn_b,
                            const float* __restrict__ bn_m, const float* __restrict__ bn_v,
                            float* __restrict__ C, int M, int K, int Nc) {
  int wave = (blockIdx.x * blockDim.x + threadIdx.x) >> 5;
  int lane = threadIdx.x & 31;
  int groupsPerRow = Nc >> 6;                 // 64-wide column groups
  int nWaves = (M >> 4) * groupsPerRow;
  if (wave >= nWaves) return;                 // wave-uniform: EXEC stays all-ones
  int tr = wave / groupsPerRow, tg = wave % groupsPerRow;
  int tilesPerRow = Nc >> 4;
  int m16  = lane & 15;
  int half = lane >> 4;
  const float* Arow = A + (size_t)(tr * 16 + m16) * K;
  const v16h*  Bv   = (const v16h*)Bp;

  v8f acc0 = {}, acc1 = {}, acc2 = {}, acc3 = {};
  for (int kk = 0; kk < K; kk += 32) {
    // A fragment: two 32-byte contiguous chunks
    const float4* p0 = (const float4*)(Arow + kk + 8 * half);
    const float4* p1 = (const float4*)(Arow + kk + 16 + 8 * half);
    float4 c0 = p0[0], c1 = p0[1];
    float4 c2 = p1[0], c3 = p1[1];
    if (kk + 32 < K) __builtin_prefetch(Arow + kk + 32 + 8 * half, 0, 1);
    v16h a;
    a[0]  = (_Float16)c0.x; a[1]  = (_Float16)c0.y; a[2]  = (_Float16)c0.z; a[3]  = (_Float16)c0.w;
    a[4]  = (_Float16)c1.x; a[5]  = (_Float16)c1.y; a[6]  = (_Float16)c1.z; a[7]  = (_Float16)c1.w;
    a[8]  = (_Float16)c2.x; a[9]  = (_Float16)c2.y; a[10] = (_Float16)c2.z; a[11] = (_Float16)c2.w;
    a[12] = (_Float16)c3.x; a[13] = (_Float16)c3.y; a[14] = (_Float16)c3.z; a[15] = (_Float16)c3.w;

    size_t bbase = ((size_t)(kk >> 5) * tilesPerRow + (tg << 2)) * 32 + lane;
    v16h b0 = Bv[bbase];
    v16h b1 = Bv[bbase + 32];
    v16h b2 = Bv[bbase + 64];
    v16h b3 = Bv[bbase + 96];
    acc0 = __builtin_amdgcn_wmma_f32_16x16x32_f16(false, a, false, b0, (short)0, acc0, false, false);
    acc1 = __builtin_amdgcn_wmma_f32_16x16x32_f16(false, a, false, b1, (short)0, acc1, false, false);
    acc2 = __builtin_amdgcn_wmma_f32_16x16x32_f16(false, a, false, b2, (short)0, acc2, false, false);
    acc3 = __builtin_amdgcn_wmma_f32_16x16x32_f16(false, a, false, b3, (short)0, acc3, false, false);
  }
  // C/D layout: lane n = lane%16 -> column; VGPR r -> row r + 8*half
  int n = lane & 15;
  #pragma unroll
  for (int q = 0; q < 4; ++q) {
    v8f accq = (q == 0) ? acc0 : (q == 1) ? acc1 : (q == 2) ? acc2 : acc3;
    int colg = ((tg << 2) + q) * 16 + n;
    float bs = bias ? bias[colg] : 0.0f;
    #pragma unroll
    for (int r = 0; r < 8; ++r) {
      int rowg = tr * 16 + r + 8 * half;
      float t = accq[r] + bs;
      if (EPI == 2)
        t = (t - bn_m[colg]) * rsqrtf(bn_v[colg] + EPSV) * bn_g[colg] + bn_b[colg];
      if (EPI >= 1) t = fmaxf(t, 0.0f);
      C[(size_t)rowg * Nc + colg] = t;
    }
  }
}

// ---------------- edge pass 1: alpha + segment max (wave per edge) ----------------
__global__ void k_edge_alpha(const int* __restrict__ src, const int* __restrict__ dst,
                             const float* __restrict__ xl, const float* __restrict__ xr,
                             const float* __restrict__ eattr, const float* __restrict__ loopA,
                             const float* __restrict__ We,   // [16,256] f32
                             const float* __restrict__ att,  // [4*64] f32
                             float* __restrict__ alpha, unsigned* __restrict__ amaxKey) {
  int wave = (blockIdx.x * blockDim.x + threadIdx.x) >> 5;
  int lane = threadIdx.x & 31;
  if (wave >= E2TOT) return;
  int e = wave;
  int s = (e < NEDGE) ? src[e] : (e - NEDGE);
  int d = (e < NEDGE) ? dst[e] : (e - NEDGE);
  const float* ea = (e < NEDGE) ? (eattr + (size_t)e * FEDGE)
                                : (loopA + (size_t)(e - NEDGE) * FEDGE);
  float eav[FEDGE];
  #pragma unroll
  for (int k = 0; k < FEDGE; ++k) eav[k] = ea[k];

  const float* xls = xl + (size_t)s * HID;
  const float* xrd = xr + (size_t)d * HID;
  float partial = 0.0f;
  int jbase = lane * 8;                      // 8 contiguous channels within one head
  #pragma unroll
  for (int t = 0; t < 8; ++t) {
    int j = jbase + t;
    float ef = 0.0f;
    #pragma unroll
    for (int k = 0; k < FEDGE; ++k) ef += eav[k] * We[k * HID + j];
    float z = xls[j] + xrd[j] + ef;
    z = (z > 0.0f) ? z : 0.2f * z;           // leaky_relu(0.2)
    partial += z * att[j];
  }
  // reduce over the 8 lanes of each head (lanes 8h..8h+7)
  #pragma unroll
  for (int off = 4; off >= 1; off >>= 1) partial += __shfl_xor(partial, off, 32);
  if ((lane & 7) == 0) {
    int hh = lane >> 3;
    alpha[(size_t)e * 4 + hh] = partial;
    atomicMax(&amaxKey[d * 4 + hh], f2key(partial));
  }
}

// ---------------- edge pass 2: exp + segment sum ----------------
__global__ void k_edge_soft(const int* __restrict__ dst, float* __restrict__ alpha,
                            const unsigned* __restrict__ amaxKey, float* __restrict__ den) {
  int idx = blockIdx.x * blockDim.x + threadIdx.x;
  if (idx >= E2TOT * 4) return;
  int e = idx >> 2, hh = idx & 3;
  int d = (e < NEDGE) ? dst[e] : (e - NEDGE);
  float ex = __expf(alpha[idx] - key2f(amaxKey[d * 4 + hh]));
  alpha[idx] = ex;                           // in-place: alpha -> ex
  atomicAdd(&den[d * 4 + hh], ex);
}

// ---------------- edge pass 3: weighted aggregation (wave per edge) ----------------
__global__ void k_edge_agg(const int* __restrict__ src, const int* __restrict__ dst,
                           const float* __restrict__ ex, const float* __restrict__ den,
                           const float* __restrict__ xl, float* __restrict__ agg) {
  int wave = (blockIdx.x * blockDim.x + threadIdx.x) >> 5;
  int lane = threadIdx.x & 31;
  if (wave >= E2TOT) return;
  int e = wave;
  int s = (e < NEDGE) ? src[e] : (e - NEDGE);
  int d = (e < NEDGE) ? dst[e] : (e - NEDGE);
  int hh = lane >> 3;
  float w = ex[(size_t)e * 4 + hh] / den[d * 4 + hh];
  const float* xls = xl + (size_t)s * HID;
  float* od = agg + (size_t)d * HID;
  int jbase = lane * 8;
  #pragma unroll
  for (int t = 0; t < 8; ++t) {
    int j = jbase + t;
    atomicAdd(&od[j], w * xls[j]);
  }
}

// ---------------- node update: BN + relu + residual ----------------
__global__ void k_node_update(const float* __restrict__ agg, const float* __restrict__ cbias,
                              const float* __restrict__ g, const float* __restrict__ b,
                              const float* __restrict__ m, const float* __restrict__ v,
                              const float* __restrict__ h_in, float* __restrict__ h_out) {
  int idx = blockIdx.x * blockDim.x + threadIdx.x;
  if (idx >= NNODE * HID) return;
  int f = idx & (HID - 1);
  float t = agg[idx] + cbias[f];
  t = (t - m[f]) * rsqrtf(v[f] + EPSV) * g[f] + b[f];
  h_out[idx] = fmaxf(t, 0.0f) + h_in[idx];
}

// ---------------- pooling ----------------
__global__ void k_pool(const float* __restrict__ h, const int* __restrict__ batch,
                       float* __restrict__ psum, unsigned* __restrict__ pmaxKey,
                       float* __restrict__ pcnt) {
  int idx = blockIdx.x * blockDim.x + threadIdx.x;
  if (idx >= NNODE * HID) return;
  int n = idx >> 8, f = idx & (HID - 1);
  int g = batch[n];
  float v = h[idx];
  atomicAdd(&psum[(size_t)g * HID + f], v);
  atomicMax(&pmaxKey[(size_t)g * HID + f], f2key(v));
  if (f == 0) atomicAdd(&pcnt[g], 1.0f);
}
__global__ void k_build_xp(const float* __restrict__ psum, const unsigned* __restrict__ pmaxKey,
                           const float* __restrict__ pcnt, float* __restrict__ xp) {
  int idx = blockIdx.x * blockDim.x + threadIdx.x;
  if (idx >= NGRAPH * 2 * HID) return;
  int g = idx / (2 * HID), f = idx % (2 * HID);
  xp[idx] = (f < HID) ? psum[(size_t)g * HID + f] / fmaxf(pcnt[g], 1.0f)
                      : key2f(pmaxKey[(size_t)g * HID + (f - HID)]);
}

// ---------------- final classifier layer ----------------
__global__ void k_cls_final(const float* __restrict__ z2, const float* __restrict__ W3,
                            const float* __restrict__ b3, float* __restrict__ out) {
  int g = blockIdx.x * blockDim.x + threadIdx.x;
  if (g >= NGRAPH) return;
  float acc = b3[0];
  #pragma unroll 8
  for (int k = 0; k < 128; ++k) acc += z2[(size_t)g * 128 + k] * W3[k];
  out[g] = acc;
}

// ===================================================================
extern "C" void kernel_launch(void* const* d_in, const int* in_sizes, int n_in,
                              void* d_out, int out_size, void* d_ws, size_t ws_size,
                              hipStream_t stream) {
  const float* x          = (const float*)d_in[0];
  const int*   edge_index = (const int*)  d_in[1];
  const int*   batch      = (const int*)  d_in[2];
  const float* edge_attr  = (const float*)d_in[3];
  const float* enc_W      = (const float*)d_in[4];
  const float* enc_b      = (const float*)d_in[5];
  const float* conv_Wl    = (const float*)d_in[6];
  const float* conv_bl    = (const float*)d_in[7];
  const float* conv_Wr    = (const float*)d_in[8];
  const float* conv_br    = (const float*)d_in[9];
  const float* conv_We    = (const float*)d_in[10];
  const float* conv_att   = (const float*)d_in[11];
  const float* conv_bias  = (const float*)d_in[12];
  const float* bn_gamma   = (const float*)d_in[13];
  const float* bn_beta    = (const float*)d_in[14];
  const float* bn_mean    = (const float*)d_in[15];
  const float* bn_var     = (const float*)d_in[16];
  const float* cls_W1     = (const float*)d_in[17];
  const float* cls_b1     = (const float*)d_in[18];
  const float* cls_g1     = (const float*)d_in[19];
  const float* cls_bt1    = (const float*)d_in[20];
  const float* cls_m1     = (const float*)d_in[21];
  const float* cls_v1     = (const float*)d_in[22];
  const float* cls_W2     = (const float*)d_in[23];
  const float* cls_b2     = (const float*)d_in[24];
  const float* cls_g2     = (const float*)d_in[25];
  const float* cls_bt2    = (const float*)d_in[26];
  const float* cls_m2     = (const float*)d_in[27];
  const float* cls_v2     = (const float*)d_in[28];
  const float* cls_W3     = (const float*)d_in[29];
  const float* cls_b3     = (const float*)d_in[30];
  float* out = (float*)d_out;

  const int* srcA = edge_index;
  const int* dstA = edge_index + NEDGE;

  // ---- workspace carve ----
  char* p = (char*)d_ws;
  auto alloc = [&](size_t bytes) -> char* {
    char* r = p; p += (bytes + 255) & ~(size_t)255; return r;
  };
  float*    h0      = (float*)   alloc((size_t)NNODE * HID * 4);
  float*    h1      = (float*)   alloc((size_t)NNODE * HID * 4);
  float*    xl      = (float*)   alloc((size_t)NNODE * HID * 4);
  float*    xr      = (float*)   alloc((size_t)NNODE * HID * 4);
  float*    agg     = (float*)   alloc((size_t)NNODE * HID * 4);
  float*    alpha   = (float*)   alloc((size_t)E2TOT * 4 * 4);
  unsigned* amaxKey = (unsigned*)alloc((size_t)NNODE * 4 * 4);
  float*    den     = (float*)   alloc((size_t)NNODE * 4 * 4);
  float*    deg     = (float*)   alloc((size_t)NNODE * 4);
  float*    loopA   = (float*)   alloc((size_t)NNODE * FEDGE * 4);
  _Float16* encW16  = (_Float16*)alloc((size_t)FNODE * HID * 2);
  _Float16* Wl16    = (_Float16*)alloc((size_t)3 * HID * HID * 2);
  _Float16* Wr16    = (_Float16*)alloc((size_t)3 * HID * HID * 2);
  _Float16* W1_16   = (_Float16*)alloc((size_t)2 * HID * HID * 2);
  _Float16* W2_16   = (_Float16*)alloc((size_t)HID * 128 * 2);
  float*    psum    = (float*)   alloc((size_t)NGRAPH * HID * 4);
  unsigned* pmaxKey = (unsigned*)alloc((size_t)NGRAPH * HID * 4);
  float*    pcnt    = (float*)   alloc((size_t)NGRAPH * 4);
  float*    xp      = (float*)   alloc((size_t)NGRAPH * 2 * HID * 4);
  float*    z1      = (float*)   alloc((size_t)NGRAPH * HID * 4);
  float*    z2      = (float*)   alloc((size_t)NGRAPH * 128 * 4);

  auto cdiv = [](int a, int b) { return (a + b - 1) / b; };
  const int TB = 256;

  // ---- weight conversion + pack into WMMA fragment order ----
  k_pack_b<<<cdiv(FNODE * HID, TB), TB, 0, stream>>>(enc_W, encW16, FNODE, HID);
  for (int i = 0; i < 3; ++i) {
    k_pack_b<<<cdiv(HID * HID, TB), TB, 0, stream>>>(
        conv_Wl + (size_t)i * HID * HID, Wl16 + (size_t)i * HID * HID, HID, HID);
    k_pack_b<<<cdiv(HID * HID, TB), TB, 0, stream>>>(
        conv_Wr + (size_t)i * HID * HID, Wr16 + (size_t)i * HID * HID, HID, HID);
  }
  k_pack_b<<<cdiv(2 * HID * HID, TB), TB, 0, stream>>>(cls_W1, W1_16, 2 * HID, HID);
  k_pack_b<<<cdiv(HID * 128, TB), TB, 0, stream>>>(cls_W2, W2_16, HID, 128);

  // ---- self-loop edge attr: mean of incoming ----
  k_fill_f32<<<cdiv(NNODE, TB), TB, 0, stream>>>(deg, 0.0f, NNODE);
  k_fill_f32<<<cdiv(NNODE * FEDGE, TB), TB, 0, stream>>>(loopA, 0.0f, NNODE * FEDGE);
  k_loop_acc<<<cdiv(NEDGE, TB), TB, 0, stream>>>(dstA, edge_attr, deg, loopA, NEDGE);
  k_loop_div<<<cdiv(NNODE * FEDGE, TB), TB, 0, stream>>>(loopA, deg, NNODE * FEDGE);

  // ---- encoder: h = relu(x @ enc_W + enc_b) ----
  {
    int waves = (NNODE / 16) * (HID / 64);
    k_gemm_wmma<1><<<cdiv(waves, 8), TB, 0, stream>>>(
        x, encW16, enc_b, nullptr, nullptr, nullptr, nullptr, h0, NNODE, FNODE, HID);
  }

  float* cur = h0;
  float* nxt = h1;
  int edgeBlocks = cdiv(E2TOT, 8);   // 8 waves per 256-thread block, wave per edge

  for (int i = 0; i < 3; ++i) {
    const _Float16* Wl_i = Wl16 + (size_t)i * HID * HID;
    const _Float16* Wr_i = Wr16 + (size_t)i * HID * HID;
    int waves = (NNODE / 16) * (HID / 64);
    k_gemm_wmma<0><<<cdiv(waves, 8), TB, 0, stream>>>(
        cur, Wl_i, conv_bl + i * HID, nullptr, nullptr, nullptr, nullptr, xl, NNODE, HID, HID);
    k_gemm_wmma<0><<<cdiv(waves, 8), TB, 0, stream>>>(
        cur, Wr_i, conv_br + i * HID, nullptr, nullptr, nullptr, nullptr, xr, NNODE, HID, HID);

    k_fill_u32<<<cdiv(NNODE * 4, TB), TB, 0, stream>>>(amaxKey, NEGINF_KEY, NNODE * 4);
    k_fill_f32<<<cdiv(NNODE * 4, TB), TB, 0, stream>>>(den, 0.0f, NNODE * 4);
    k_fill_f32<<<cdiv(NNODE * HID, TB), TB, 0, stream>>>(agg, 0.0f, NNODE * HID);

    k_edge_alpha<<<edgeBlocks, TB, 0, stream>>>(
        srcA, dstA, xl, xr, edge_attr, loopA,
        conv_We + (size_t)i * FEDGE * HID, conv_att + (size_t)i * HID, alpha, amaxKey);
    k_edge_soft<<<cdiv(E2TOT * 4, TB), TB, 0, stream>>>(dstA, alpha, amaxKey, den);
    k_edge_agg<<<edgeBlocks, TB, 0, stream>>>(srcA, dstA, alpha, den, xl, agg);

    k_node_update<<<cdiv(NNODE * HID, TB), TB, 0, stream>>>(
        agg, conv_bias + i * HID, bn_gamma + i * HID, bn_beta + i * HID,
        bn_mean + i * HID, bn_var + i * HID, cur, nxt);
    float* t = cur; cur = nxt; nxt = t;
  }

  // ---- pooling (mean | max) ----
  k_fill_f32<<<cdiv(NGRAPH * HID, TB), TB, 0, stream>>>(psum, 0.0f, NGRAPH * HID);
  k_fill_u32<<<cdiv(NGRAPH * HID, TB), TB, 0, stream>>>(pmaxKey, NEGINF_KEY, NGRAPH * HID);
  k_fill_f32<<<cdiv(NGRAPH, TB), TB, 0, stream>>>(pcnt, 0.0f, NGRAPH);
  k_pool<<<cdiv(NNODE * HID, TB), TB, 0, stream>>>(cur, batch, psum, pmaxKey, pcnt);
  k_build_xp<<<cdiv(NGRAPH * 2 * HID, TB), TB, 0, stream>>>(psum, pmaxKey, pcnt, xp);

  // ---- classifier ----
  {
    int waves = (NGRAPH / 16) * (HID / 64);
    k_gemm_wmma<2><<<cdiv(waves, 8), TB, 0, stream>>>(
        xp, W1_16, cls_b1, cls_g1, cls_bt1, cls_m1, cls_v1, z1, NGRAPH, 2 * HID, HID);
  }
  {
    int waves = (NGRAPH / 16) * (128 / 64);
    k_gemm_wmma<2><<<cdiv(waves, 8), TB, 0, stream>>>(
        z1, W2_16, cls_b2, cls_g2, cls_bt2, cls_m2, cls_v2, z2, NGRAPH, HID, 128);
  }
  k_cls_final<<<cdiv(NGRAPH, TB), TB, 0, stream>>>(z2, cls_W3, cls_b3, out);
}
